// KCDiscovery_54571854463439
// MI455X (gfx1250) — matrix operand order repaired and checked by reading, not compiled, not run on hardware
//
#include <hip/hip_runtime.h>
#include <math.h>

#define NPTS 50000
#define DIM  256
#define KC   256
#define ROWS 64
#define THREADS 128
#define NBLK ((NPTS + ROWS - 1) / ROWS)   // 782
#define KM_EPS 1e-5f
#define SM_TAU 0.243f

typedef float  v8f   __attribute__((ext_vector_type(8)));
typedef __bf16 v4bf  __attribute__((ext_vector_type(4)));
typedef __bf16 v8bf  __attribute__((ext_vector_type(8)));
typedef __bf16 v16bf __attribute__((ext_vector_type(16)));
typedef unsigned int v4u __attribute__((ext_vector_type(4)));
typedef int v4i __attribute__((ext_vector_type(4)));
typedef int v8i __attribute__((ext_vector_type(8)));

#if defined(__has_builtin)
#if __has_builtin(__builtin_amdgcn_tensor_load_to_lds) && __has_builtin(__builtin_amdgcn_s_wait_tensorcnt)
#define HAVE_TDM 1
#endif
#endif

struct bf2 { __bf16 h, l; };

__device__ __forceinline__ bf2 bsplit(float v) {
    bf2 r;
    r.h = (__bf16)v;
    r.l = (__bf16)(v - (float)r.h);
    return r;
}

__device__ __forceinline__ v16bf cat16(v8bf a, v8bf b) {
    return __builtin_shufflevector(a, b, 0,1,2,3,4,5,6,7,8,9,10,11,12,13,14,15);
}

__device__ __forceinline__ v8f wmma_bf16(v16bf a, v16bf b, v8f c) {
    // D(16x16 f32) = A(16x32 bf16) * B(32x16 bf16) + C
    return __builtin_amdgcn_wmma_f32_16x16x32_bf16(
        /*neg_a=*/false, a, /*neg_b=*/false, b,
        /*c_mod=*/(short)0, c, /*reuse_a=*/false, /*reuse_b=*/false);
}

// ---------------------------------------------------------------------------
// Fused kernel, split-bf16 WMMA GEMMs + TDM tile staging.
//   mode 0: logits -> softmax -> r ; accumulate w_k and r^T x (atomics)
//   mode 1: logits -> gumbel-softmax -> write y
// Block = 4 waves = 64 rows. LDS 144KB, phase-overlaid.
// ---------------------------------------------------------------------------
__global__ __launch_bounds__(THREADS) void kc_main(
    const float* __restrict__ x, const float* __restrict__ cw,
    const float* __restrict__ ccb, const float* __restrict__ logtau,
    float* __restrict__ accg, float* __restrict__ waccg,
    const float* __restrict__ noise, float* __restrict__ out, int finalmode)
{
    __shared__ float  lsf[ROWS * DIM];       // 64KB: f32 x tile / logits ; then xT hi/lo
    __shared__ __bf16 xa[2 * ROWS * DIM];    // 64KB: x hi/lo  ; then rT hi/lo
    __shared__ __bf16 ca[2 * 16 * DIM];      // 16KB: centroid tile hi/lo

    __bf16* xsh = xa;                        // [64][256] row-major
    __bf16* xsl = xa + ROWS * DIM;
    __bf16* csh = ca;                        // [16][256]
    __bf16* csl = ca + 16 * DIM;
    float*  ls  = lsf;                       // [64][256] f32
    __bf16* rTh = xa;                        // [256 k][64 row]  (phase 2)
    __bf16* rTl = xa + ROWS * DIM;
    __bf16* xTh = (__bf16*)lsf;              // [256 d][64 row]  (phase 2)
    __bf16* xTl = ((__bf16*)lsf) + ROWS * DIM;

    const int tid   = threadIdx.x;
    const int wid   = tid >> 5;              // wave 0..3 (wave32)
    const int lane  = tid & 31;
    const int half  = lane >> 4;
    const int l15   = lane & 15;
    const int rbase = blockIdx.x * ROWS;
    const float invtau = __expf(-logtau[0]);

#ifdef HAVE_TDM
    // ---- TDM: DMA the 64x256 f32 x tile into LDS (lsf). OOB rows read 0. ----
    if (wid == 0) {
        const unsigned ldsoff = (unsigned)(uintptr_t)&lsf[0];
        const unsigned long long ga =
            (unsigned long long)(uintptr_t)(x + (size_t)rbase * DIM);
        const unsigned rows_rem = (unsigned)(NPTS - rbase);   // tensor dim1 (OOB mask)
        v4u g0;
        g0[0] = 1u;                                           // count=1, user mode
        g0[1] = ldsoff;                                       // lds_addr
        g0[2] = (unsigned)ga;                                 // global_addr[31:0]
        g0[3] = ((unsigned)(ga >> 32) & 0x01FFFFFFu)          // global_addr[56:32]
                | 0x80000000u;                                // type=2 ("image")
        v8i g1;
        g1[0] = 0x20000;                        // data_size=2 (4B), no multicast/pad
        g1[1] = (int)(256u << 16);              // tensor_dim0 = 256
        g1[2] = (int)((rows_rem & 0xFFFFu) << 16); // tensor_dim1 = rows remaining
        g1[3] = (int)(256u << 16);              // tile_dim0 = 256
        g1[4] = ROWS;                           // tile_dim1 = 64, tile_dim2 = 0
        g1[5] = 256;                            // tensor_dim0_stride = 256 elements
        g1[6] = 0;
        g1[7] = 0;
        v4i gz4 = {0, 0, 0, 0};                 // groups 2/3 unused (2D tensor)
        v8i gz8 = {0, 0, 0, 0, 0, 0, 0, 0};
        __builtin_amdgcn_tensor_load_to_lds(g0, g1, gz4, gz4, gz8, 0);
        __builtin_amdgcn_s_wait_tensorcnt(0);
    }
    __syncthreads();
    // convert f32 tile (LDS) -> split bf16 operands
    for (int i = tid; i < ROWS * DIM / 4; i += THREADS) {
        float4 f = *(const float4*)&lsf[i * 4];
        v4bf h, l;
        bf2 s0 = bsplit(f.x), s1 = bsplit(f.y), s2 = bsplit(f.z), s3 = bsplit(f.w);
        h[0] = s0.h; h[1] = s1.h; h[2] = s2.h; h[3] = s3.h;
        l[0] = s0.l; l[1] = s1.l; l[2] = s2.l; l[3] = s3.l;
        *(v4bf*)&xsh[i * 4] = h;
        *(v4bf*)&xsl[i * 4] = l;
    }
    __syncthreads();
#else
    // ---- fallback: stage x tile as split bf16 via coalesced global reads ----
    {
        const float4* x4 = (const float4*)x;
        for (int i = tid; i < ROWS * DIM / 4; i += THREADS) {
            int row = rbase + (i >> 6);
            float4 f = (row < NPTS) ? x4[(size_t)rbase * (DIM / 4) + i]
                                    : make_float4(0.f, 0.f, 0.f, 0.f);
            v4bf h, l;
            bf2 s0 = bsplit(f.x), s1 = bsplit(f.y), s2 = bsplit(f.z), s3 = bsplit(f.w);
            h[0] = s0.h; h[1] = s1.h; h[2] = s2.h; h[3] = s3.h;
            l[0] = s0.l; l[1] = s1.l; l[2] = s2.l; l[3] = s3.l;
            *(v4bf*)&xsh[i * 4] = h;
            *(v4bf*)&xsl[i * 4] = l;
        }
    }
    __syncthreads();
#endif

    // ---- GEMM1: S = x * c^T (split bf16: xh*ch + xh*cl + xl*ch) ----
    for (int nt = 0; nt < 16; ++nt) {
        {
            const float4* c4 = (const float4*)cw;
            for (int i = tid; i < 16 * DIM / 4; i += THREADS) {
                float4 f = c4[nt * 16 * (DIM / 4) + i];
                v4bf h, l;
                bf2 s0 = bsplit(f.x), s1 = bsplit(f.y), s2 = bsplit(f.z), s3 = bsplit(f.w);
                h[0] = s0.h; h[1] = s1.h; h[2] = s2.h; h[3] = s3.h;
                l[0] = s0.l; l[1] = s1.l; l[2] = s2.l; l[3] = s3.l;
                *(v4bf*)&csh[i * 4] = h;
                *(v4bf*)&csl[i * 4] = l;
            }
        }
        __syncthreads();

        v8f acc = {};
        const __bf16* arh = &xsh[(wid * 16 + l15) * DIM];
        const __bf16* arl = &xsl[(wid * 16 + l15) * DIM];
        const __bf16* brh = &csh[l15 * DIM];
        const __bf16* brl = &csl[l15 * DIM];
        #pragma unroll
        for (int ks = 0; ks < DIM / 32; ++ks) {
            const int ao = ks * 32 + half * 8;    // A: K runs [ao..ao+7],[ao+16..ao+23]
            const int bo = ks * 32 + half * 8;    // B: K runs [bo..bo+7],[bo+16..bo+23]
            v16bf ah = cat16(*(const v8bf*)(arh + ao), *(const v8bf*)(arh + ao + 16));
            v16bf al = cat16(*(const v8bf*)(arl + ao), *(const v8bf*)(arl + ao + 16));
            v16bf bh = cat16(*(const v8bf*)(brh + bo), *(const v8bf*)(brh + bo + 16));
            v16bf bl = cat16(*(const v8bf*)(brl + bo), *(const v8bf*)(brl + bo + 16));
            acc = wmma_bf16(ah, bh, acc);
            acc = wmma_bf16(ah, bl, acc);
            acc = wmma_bf16(al, bh, acc);
        }
        #pragma unroll
        for (int j = 0; j < 8; ++j)
            ls[(wid * 16 + j + 8 * half) * DIM + nt * 16 + l15] = acc[j];
        __syncthreads();
    }

    // ---- per-row softmax over K (||x||^2 cancels). mode0 writes rT hi/lo ----
    const float istau = finalmode ? (1.0f / SM_TAU) : 1.0f;
    if (tid < ROWS) {
        const int grow = rbase + tid;
        float* lr = &ls[tid * DIM];
        if (grow >= NPTS) {
            if (!finalmode)
                for (int k = 0; k < KC; ++k) { rTh[k * ROWS + tid] = (__bf16)0.f;
                                               rTl[k * ROWS + tid] = (__bf16)0.f; }
        } else {
            if (finalmode) __builtin_prefetch(&noise[(size_t)grow * KC], 0, 1);
            float mx = -3.0e38f;
            for (int k = 0; k < KC; ++k) {
                float v = (2.0f * lr[k] - ccb[k]) * invtau * istau;
                if (finalmode) {
                    float u = noise[(size_t)grow * KC + k];
                    v += -__logf(-__logf(u));
                }
                lr[k] = v;
                mx = fmaxf(mx, v);
            }
            float s = 0.0f;
            for (int k = 0; k < KC; ++k) { float e = __expf(lr[k] - mx); lr[k] = e; s += e; }
            float inv = 1.0f / s;
            if (finalmode) {
                for (int k = 0; k < KC; ++k) lr[k] *= inv;
            } else {
                for (int k = 0; k < KC; ++k) {
                    bf2 s2 = bsplit(lr[k] * inv);
                    rTh[k * ROWS + tid] = s2.h;
                    rTl[k * ROWS + tid] = s2.l;
                }
            }
        }
    }
    __syncthreads();

    if (finalmode) {   // stream y out, coalesced
        float4* o4 = (float4*)out;
        const float4* l4 = (const float4*)ls;
        for (int i = tid; i < ROWS * DIM / 4; i += THREADS) {
            int row = rbase + (i >> 6);
            if (row < NPTS) o4[(size_t)rbase * (DIM / 4) + i] = l4[i];
        }
        return;
    }

    // ---- rebuild x tile transposed (split bf16) into dead logits region ----
    for (int i = tid; i < ROWS * DIM; i += THREADS) {
        int row = i >> 8, d = i & (DIM - 1);
        float f = (rbase + row < NPTS) ? x[(size_t)rbase * DIM + i] : 0.f;
        bf2 s = bsplit(f);
        xTh[d * ROWS + row] = s.h;
        xTl[d * ROWS + row] = s.l;
    }
    __syncthreads();

    // ---- w_k += sum_rows r (from split rT, consistent with GEMM2) ----
    {
        #pragma unroll
        for (int kk = 0; kk < 2; ++kk) {
            int k = tid + kk * THREADS;
            float s = 0.f;
            for (int r = 0; r < ROWS; ++r)
                s += (float)rTh[k * ROWS + r] + (float)rTl[k * ROWS + r];
            atomicAdd(&waccg[k], s);
        }
    }

    // ---- GEMM2: acc[k][d] += r^T x  (rh*xh + rh*xl + rl*xh) ----
    for (int kt = 0; kt < 4; ++kt) {
        const int kcol = (wid * 4 + kt) * 16 + l15;     // centroid (M)
        const __bf16* arh = &rTh[kcol * ROWS];
        const __bf16* arl = &rTl[kcol * ROWS];
        for (int dt = 0; dt < 16; ++dt) {
            const int dcol = dt * 16 + l15;             // dim (N)
            const __bf16* brh = &xTh[dcol * ROWS];
            const __bf16* brl = &xTl[dcol * ROWS];
            v8f acc = {};
            #pragma unroll
            for (int ks = 0; ks < ROWS / 32; ++ks) {
                const int ao = ks * 32 + half * 8;
                const int bo = ks * 32 + half * 8;
                v16bf ah = cat16(*(const v8bf*)(arh + ao), *(const v8bf*)(arh + ao + 16));
                v16bf al = cat16(*(const v8bf*)(arl + ao), *(const v8bf*)(arl + ao + 16));
                v16bf bh = cat16(*(const v8bf*)(brh + bo), *(const v8bf*)(brh + bo + 16));
                v16bf bl = cat16(*(const v8bf*)(brl + bo), *(const v8bf*)(brl + bo + 16));
                acc = wmma_bf16(ah, bh, acc);
                acc = wmma_bf16(ah, bl, acc);
                acc = wmma_bf16(al, bh, acc);
            }
            #pragma unroll
            for (int j = 0; j < 8; ++j) {
                int kc = (wid * 4 + kt) * 16 + j + 8 * half;
                atomicAdd(&accg[kc * DIM + dcol], acc[j]);
            }
        }
    }
}

// ---------------------------------------------------------------------------
__global__ __launch_bounds__(256) void kc_init(
    const float* __restrict__ cin, float* __restrict__ cw,
    float* __restrict__ ccb, float* __restrict__ donef)
{
    __shared__ float red[256];
    const int b = blockIdx.x, t = threadIdx.x;
    float v = cin[b * DIM + t];
    cw[b * DIM + t] = v;
    red[t] = v * v;
    __syncthreads();
    for (int s = 128; s > 0; s >>= 1) { if (t < s) red[t] += red[t + s]; __syncthreads(); }
    if (t == 0) ccb[b] = red[0];
    if (b == 0 && t == 0) { donef[0] = 0.f; donef[1] = 0.f; }
}

__global__ __launch_bounds__(256) void kc_zero(
    float* __restrict__ accg, float* __restrict__ waccg, float* __restrict__ shift2)
{
    const int idx = blockIdx.x * 256 + threadIdx.x;
    accg[idx] = 0.f;
    if (idx < KC) waccg[idx] = 0.f;
    if (idx == 0) *shift2 = 0.f;
}

__global__ __launch_bounds__(256) void kc_reduce(
    const float* __restrict__ accg, const float* __restrict__ waccg,
    const float* __restrict__ cw, float* __restrict__ newc, float* __restrict__ shift2)
{
    __shared__ float red[256];
    const int k = blockIdx.x, t = threadIdx.x;
    float w = waccg[k] + 1e-8f;
    float nc = accg[k * DIM + t] / w;
    newc[k * DIM + t] = nc;
    float d = nc - cw[k * DIM + t];
    red[t] = d * d;
    __syncthreads();
    for (int s = 128; s > 0; s >>= 1) { if (t < s) red[t] += red[t + s]; __syncthreads(); }
    if (t == 0) atomicAdd(shift2, red[0]);
}

__global__ __launch_bounds__(256) void kc_finalize(
    float* __restrict__ cw, const float* __restrict__ newc, float* __restrict__ ccb,
    const float* __restrict__ shift2, float* __restrict__ donef, int parity)
{
    __shared__ float red[256];
    const int b = blockIdx.x, t = threadIdx.x;
    if (b < KC) {
        float dold = donef[parity];                  // OLD done flag (ref semantics)
        float v = (dold != 0.0f) ? cw[b * DIM + t] : newc[b * DIM + t];
        cw[b * DIM + t] = v;
        red[t] = v * v;
        __syncthreads();
        for (int s = 128; s > 0; s >>= 1) { if (t < s) red[t] += red[t + s]; __syncthreads(); }
        if (t == 0) ccb[b] = red[0];
    } else if (t == 0) {
        float dold = donef[parity];
        donef[parity ^ 1] = (dold != 0.0f || *shift2 < KM_EPS * KM_EPS) ? 1.f : 0.f;
    }
}

// ---------------------------------------------------------------------------
extern "C" void kernel_launch(void* const* d_in, const int* in_sizes, int n_in,
                              void* d_out, int out_size, void* d_ws, size_t ws_size,
                              hipStream_t stream) {
    (void)in_sizes; (void)n_in; (void)out_size; (void)ws_size;
    const float* x      = (const float*)d_in[0];   // [N, D]
    const float* cin    = (const float*)d_in[1];   // [K, D]
    const float* logtau = (const float*)d_in[2];   // [1]
    const float* noise  = (const float*)d_in[3];   // [N, K]
    float* out = (float*)d_out;                    // [N, K]

    float* w      = (float*)d_ws;
    float* cw     = w;             // 65536 : current centroids
    float* accg   = w + 65536;     // 65536 : r^T x accumulator
    float* newc   = w + 131072;    // 65536 : updated centroids
    float* ccb    = w + 196608;    //   256 : ||c_k||^2
    float* waccg  = w + 196864;    //   256 : w_k
    float* shift2 = w + 197120;    //     1 : ||dc||^2
    float* donef  = w + 197121;    //     2 : done flags (ping-pong)

    kc_init<<<KC, 256, 0, stream>>>(cin, cw, ccb, donef);
    for (int it = 0; it < 10; ++it) {
        kc_zero<<<KC, 256, 0, stream>>>(accg, waccg, shift2);
        kc_main<<<NBLK, THREADS, 0, stream>>>(x, cw, ccb, logtau, accg, waccg,
                                              noise, out, 0);
        kc_reduce<<<KC, 256, 0, stream>>>(accg, waccg, cw, newc, shift2);
        kc_finalize<<<KC + 1, 256, 0, stream>>>(cw, newc, ccb, shift2, donef, it & 1);
    }
    kc_main<<<NBLK, THREADS, 0, stream>>>(x, cw, ccb, logtau, accg, waccg,
                                          noise, out, 1);
}
